// RNN_Build_74869869904700
// MI455X (gfx1250) — matrix operand order repaired and checked by reading, not compiled, not run on hardware
//
#include <hip/hip_runtime.h>

// ---------------- CDNA5 WMMA types ----------------
typedef __bf16 bf16_t;
typedef bf16_t v16bf __attribute__((ext_vector_type(16)));
typedef float  v8f   __attribute__((ext_vector_type(8)));
typedef unsigned int v4u __attribute__((ext_vector_type(4)));

union Frag { v16bf v; v4u q[2]; };

__device__ __forceinline__ v8f wmma_bf16(v16bf a, v16bf b, v8f c) {
  return __builtin_amdgcn_wmma_f32_16x16x32_bf16(false, a, false, b, (short)0, c,
                                                 false, false);
}

// ---------------- bf16 helpers (round-to-nearest-even) ----------------
__device__ __forceinline__ unsigned int rnd_bf16(unsigned int u) {
  return u + 0x7FFFu + ((u >> 16) & 1u);
}
__device__ __forceinline__ unsigned short f2bf(float x) {
  return (unsigned short)(rnd_bf16(__float_as_uint(x)) >> 16);
}
__device__ __forceinline__ unsigned int pack2_bf16(float a, float b) {
  return (rnd_bf16(__float_as_uint(a)) >> 16) |
         (rnd_bf16(__float_as_uint(b)) & 0xFFFF0000u);
}
__device__ __forceinline__ float bf2f(unsigned short s) {
  return __uint_as_float(((unsigned int)s) << 16);
}

// ---------------- fragment loaders ----------------
// A: 16x32 bf16 tile from row-major [16][512] (LDS). Lane l<16 row M=l holds
// K = kk+half*8 .. +7 (VGPR0-3) and K = kk+16+half*8 .. +7 (VGPR4-7).
__device__ __forceinline__ v16bf load_a_frag(const unsigned short* base, int lane,
                                             int kk) {
  const int l = lane & 15, h = lane >> 4;
  const unsigned short* p = base + l * 512 + kk + (h << 3);
  Frag f;
  f.q[0] = *(const v4u*)p;
  f.q[1] = *(const v4u*)(p + 16);
  return f.v;
}
// B: 32x16 bf16 tile from pre-transposed weights WT[n][k] (k contiguous, K=512).
// Lane l column N=l%16 holds K = kk+half*8 .. +7 (VGPR0-3) and K = kk+16+half*8 .. +7 (VGPR4-7).
__device__ __forceinline__ v16bf load_b_frag(const unsigned short* wt, int lane,
                                             int n0, int kk) {
  const int l = lane & 15, h = lane >> 4;
  const unsigned short* p = wt + ((size_t)(n0 + l) << 9) + kk + (h << 3);
  Frag f;
  f.q[0] = *(const v4u*)p;
  f.q[1] = *(const v4u*)(p + 16);
  return f.v;
}

// ---------------- kernel 0: fp32 [K][N] -> bf16 transposed [N][K] ----------------
__global__ void wconvert_kernel(const float* __restrict__ src,
                                unsigned short* __restrict__ dst) {
  const int idx = blockIdx.x * 256 + threadIdx.x;  // 512*512 total
  const int n = idx >> 9, k = idx & 511;
  dst[idx] = f2bf(src[k * 512 + n]);
}

// ---------------- kernel 1: pre0[t,b,:] = emb[inputs[b,t]] @ W0x + b0 ----------
__global__ __launch_bounds__(256) void embed_gemm_kernel(
    const int* __restrict__ inp, const float* __restrict__ emb,
    const unsigned short* __restrict__ W0xT, const float* __restrict__ b0,
    float* __restrict__ pre0) {
  __shared__ alignas(16) unsigned short sA[16][512];
  __shared__ int sidx[16];
  const int tid = threadIdx.x;
  const int r0 = blockIdx.x * 16;      // row in [T*B), t-major
  const int t = r0 >> 10;              // r0 / 1024
  const int brow = r0 & 1023;
  if (tid < 16) {
    int ix = inp[(brow + tid) * 80 + t];
    if ((unsigned)ix >= 50000u) ix = 0;  // safety clamp
    sidx[tid] = ix;
  }
  __syncthreads();
  // gather 16 embedding rows -> LDS bf16
  for (int i = tid; i < 16 * 256; i += 256) {
    const int row = i >> 8, pos = i & 255;
    const float* er = emb + ((size_t)sidx[row] << 9) + (pos << 1);
    ((unsigned int*)&sA[row][0])[pos] = pack2_bf16(er[0], er[1]);
  }
  __syncthreads();

  const int lane = tid & 31, w = tid >> 5;
  const int nw = w << 6;  // 8 waves x 64-wide N slices
  const int l = lane & 15, half = lane >> 4;
  v8f z = {};
  v8f acc[4] = {z, z, z, z};
  for (int kk = 0; kk < 512; kk += 32) {
    v16bf a = load_a_frag(&sA[0][0], lane, kk);
#pragma unroll
    for (int nt = 0; nt < 4; ++nt)
      acc[nt] = wmma_bf16(a, load_b_frag(W0xT, lane, nw + nt * 16, kk), acc[nt]);
  }
#pragma unroll
  for (int nt = 0; nt < 4; ++nt) {
    const int n = nw + nt * 16 + l;
#pragma unroll
    for (int r = 0; r < 8; ++r) {
      const int m = r + (half << 3);
      pre0[((size_t)(r0 + m) << 9) + n] = acc[nt][r] + b0[n];
    }
  }
}

// ---------------- kernel 2: persistent 80-step scan, 16 batch rows / block ------
__global__ __launch_bounds__(512) void rnn_scan_kernel(
    const float* __restrict__ pre0, const unsigned short* __restrict__ W0hT,
    const unsigned short* __restrict__ W1xT, const unsigned short* __restrict__ W1hT,
    const float* __restrict__ b1, const float* __restrict__ Wout,
    const float* __restrict__ bout, float* __restrict__ out) {
  __shared__ alignas(16) unsigned short h0[16][512];
  __shared__ alignas(16) unsigned short h1[16][512];
  __shared__ float red[16][32];
  const int tid = threadIdx.x;
  const int m0 = blockIdx.x * 16;
  for (int i = tid; i < 4096; i += 512) {
    ((unsigned int*)&h0[0][0])[i] = 0u;
    ((unsigned int*)&h1[0][0])[i] = 0u;
  }
  __syncthreads();

  const int lane = tid & 31, w = tid >> 5;  // 16 waves
  const int nw = w << 5;                    // 32-wide N slice per wave
  const int l = lane & 15, half = lane >> 4;
  v8f z = {};
  float hv[2][8];

  for (int t = 0; t < 80; ++t) {
    // ---- layer 0: h0' = tanh(pre0[t] + h0 @ W0h) ----
    v8f acc[2] = {z, z};
    for (int kk = 0; kk < 512; kk += 32) {
      v16bf a = load_a_frag(&h0[0][0], lane, kk);
      acc[0] = wmma_bf16(a, load_b_frag(W0hT, lane, nw, kk), acc[0]);
      acc[1] = wmma_bf16(a, load_b_frag(W0hT, lane, nw + 16, kk), acc[1]);
    }
    const float* p0 = pre0 + ((size_t)t << 19) + ((size_t)m0 << 9);
#pragma unroll
    for (int nt = 0; nt < 2; ++nt) {
      const int n = nw + nt * 16 + l;
#pragma unroll
      for (int r = 0; r < 8; ++r) {
        const int m = r + (half << 3);
        hv[nt][r] = tanhf(acc[nt][r] + p0[((size_t)m << 9) + n]);
      }
    }
    __syncthreads();  // all waves finished reading old h0
#pragma unroll
    for (int nt = 0; nt < 2; ++nt) {
      const int n = nw + nt * 16 + l;
#pragma unroll
      for (int r = 0; r < 8; ++r) h0[r + (half << 3)][n] = f2bf(hv[nt][r]);
    }
    __syncthreads();  // new h0 visible

    // ---- layer 1: h1' = tanh(h0' @ W1x + h1 @ W1h + b1) ----
    acc[0] = z;
    acc[1] = z;
    for (int kk = 0; kk < 512; kk += 32) {
      v16bf a0 = load_a_frag(&h0[0][0], lane, kk);
      v16bf a1 = load_a_frag(&h1[0][0], lane, kk);
      acc[0] = wmma_bf16(a0, load_b_frag(W1xT, lane, nw, kk), acc[0]);
      acc[1] = wmma_bf16(a0, load_b_frag(W1xT, lane, nw + 16, kk), acc[1]);
      acc[0] = wmma_bf16(a1, load_b_frag(W1hT, lane, nw, kk), acc[0]);
      acc[1] = wmma_bf16(a1, load_b_frag(W1hT, lane, nw + 16, kk), acc[1]);
    }
#pragma unroll
    for (int nt = 0; nt < 2; ++nt) {
      const int n = nw + nt * 16 + l;
#pragma unroll
      for (int r = 0; r < 8; ++r) hv[nt][r] = tanhf(acc[nt][r] + b1[n]);
    }
    __syncthreads();  // all waves finished reading old h1
#pragma unroll
    for (int nt = 0; nt < 2; ++nt) {
      const int n = nw + nt * 16 + l;
#pragma unroll
      for (int r = 0; r < 8; ++r) h1[r + (half << 3)][n] = f2bf(hv[nt][r]);
    }
    __syncthreads();  // new h1 visible
  }

  // ---- out = sigmoid(h1_last @ Wout + bout), 32 lanes per batch row ----
  const int row = tid >> 5, sub = tid & 31;
  float s = 0.f;
  for (int j = 0; j < 16; ++j) {
    const int k = sub * 16 + j;
    s += bf2f(h1[row][k]) * Wout[k];
  }
  red[row][sub] = s;
  __syncthreads();
  if (sub == 0) {
    float tot = bout[0];
#pragma unroll
    for (int i = 0; i < 32; ++i) tot += red[row][i];
    out[m0 + row] = 1.f / (1.f + expf(-tot));
  }
}

// ---------------- host-side launch ----------------
extern "C" void kernel_launch(void* const* d_in, const int* in_sizes, int n_in,
                              void* d_out, int out_size, void* d_ws, size_t ws_size,
                              hipStream_t stream) {
  (void)in_sizes; (void)n_in; (void)out_size; (void)ws_size;
  const int* inputs  = (const int*)d_in[0];
  const float* emb   = (const float*)d_in[1];
  const float* W0x   = (const float*)d_in[2];
  const float* W0h   = (const float*)d_in[3];
  const float* b0    = (const float*)d_in[4];
  const float* W1x   = (const float*)d_in[5];
  const float* W1h   = (const float*)d_in[6];
  const float* b1    = (const float*)d_in[7];
  const float* Wout  = (const float*)d_in[8];
  const float* bout  = (const float*)d_in[9];
  float* out = (float*)d_out;

  char* ws = (char*)d_ws;
  float* pre0 = (float*)ws;  // [80][1024][512] fp32 = 160 MiB
  size_t off = (size_t)80 * 1024 * 512 * sizeof(float);
  unsigned short* W0xT = (unsigned short*)(ws + off); off += (size_t)512 * 512 * 2;
  unsigned short* W0hT = (unsigned short*)(ws + off); off += (size_t)512 * 512 * 2;
  unsigned short* W1xT = (unsigned short*)(ws + off); off += (size_t)512 * 512 * 2;
  unsigned short* W1hT = (unsigned short*)(ws + off); off += (size_t)512 * 512 * 2;

  wconvert_kernel<<<1024, 256, 0, stream>>>(W0x, W0xT);
  wconvert_kernel<<<1024, 256, 0, stream>>>(W0h, W0hT);
  wconvert_kernel<<<1024, 256, 0, stream>>>(W1x, W1xT);
  wconvert_kernel<<<1024, 256, 0, stream>>>(W1h, W1hT);
  embed_gemm_kernel<<<80 * 1024 / 16, 256, 0, stream>>>(inputs, emb, W0xT, b0, pre0);
  rnn_scan_kernel<<<1024 / 16, 512, 0, stream>>>(pre0, W0hT, W1xT, W1hT, b1, Wout,
                                                 bout, out);
}
